// CUST_40381282517287
// MI455X (gfx1250) — matrix operand order repaired and checked by reading, not compiled, not run on hardware
//
#include <hip/hip_runtime.h>
#include <hip/hip_fp16.h>

typedef __attribute__((ext_vector_type(16))) _Float16 v16h;
typedef __attribute__((ext_vector_type(8)))  _Float16 v8h;
typedef __attribute__((ext_vector_type(8)))  float    v8f;

static constexpr int Hh   = 288;
static constexpr int Wd   = 288;
static constexpr int HWp  = Hh * Wd;        // 82944
static constexpr int Bn   = 4;
static constexpr int Cd   = 64;
static constexpr int hCd  = 128;
static constexpr int TOKS = Bn * HWp;       // 331776
static constexpr int NPB  = 5184;           // tokens per (batch,group)
static constexpr int NCH  = 81;             // chunks / windows per group

// ---------------------------------------------------------------- helpers

__device__ __forceinline__ float gelu_f(float x) {
    return 0.5f * x * (1.0f + erff(x * 0.70710678118654752f));
}

__device__ __forceinline__ v8f wmma_f16(v16h a, v16h b, v8f c) {
    return __builtin_amdgcn_wmma_f32_16x16x32_f16(
        false, a, false, b, (short)0, c, false, false);
}

// A fragment (16x32 f16) from row-major buffer: lane holds row m=lane&15,
// element idx maps to K = 16*(lane>>4) + idx
__device__ __forceinline__ v16h frag_a_rm(const _Float16* row0, int stride,
                                          int k0, int lane) {
    int m = lane & 15, hf = lane >> 4;
    const _Float16* p = row0 + (size_t)m * stride + k0 + 16 * hf;
    v8h lo = *(const v8h*)p;
    v8h hi = *(const v8h*)(p + 8);
    v16h a;
#pragma unroll
    for (int i = 0; i < 8; ++i) { a[i] = lo[i]; a[i + 8] = hi[i]; }
    return a;
}

// B fragment from pre-swizzled weights: 512 halfs per (kt,nt) tile,
// lane reads 16 contiguous halfs.
__device__ __forceinline__ v16h frag_b_swz(const _Float16* w, int tile, int lane) {
    return *(const v16h*)(w + ((size_t)tile * 32 + lane) * 16);
}

// B fragment for B = X^T where X is row-major [n][K]; lane's column n is a
// row of X, K run is contiguous: K = ktile*32 + 16*(lane>>4) + idx.
__device__ __forceinline__ v16h frag_bT(const _Float16* xrow, int k0, int lane) {
    return *(const v16h*)(xrow + k0 + 16 * (lane >> 4));
}

// ---------------------------------------------------------------- kernels

// 1) x1 = x + dwconv3x3(x, pe_w, pe_b)
__global__ void k_pe(const float* __restrict__ x, const float* __restrict__ w,
                     const float* __restrict__ bias, float* __restrict__ x1) {
    int idx = blockIdx.x * 256 + threadIdx.x;
    if (idx >= Bn * Cd * HWp) return;
    int pos = idx % HWp;
    int bc  = idx / HWp;
    int c   = bc % Cd;
    int y = pos / Wd, xx = pos % Wd;
    float acc = bias[c];
    const float* base = x + (size_t)bc * HWp;
#pragma unroll
    for (int dy = 0; dy < 3; ++dy) {
        int yy = y + dy - 1;
        if ((unsigned)yy >= (unsigned)Hh) continue;
#pragma unroll
        for (int dx = 0; dx < 3; ++dx) {
            int xc = xx + dx - 1;
            if ((unsigned)xc >= (unsigned)Wd) continue;
            acc += base[yy * Wd + xc] * w[c * 9 + dy * 3 + dx];
        }
    }
    x1[idx] = x[idx] + acc;
}

// 2) weight swizzle f32 -> f16 WMMA-B layout
__global__ void k_wprep(const float* __restrict__ src, _Float16* __restrict__ dst,
                        int Ncols, int NT, int total) {
    int e = blockIdx.x * 256 + threadIdx.x;
    if (e >= total) return;
    int tile = e >> 9, ein = e & 511, lane = ein >> 4, idx = ein & 15;
    int kt = tile / NT, nt = tile % NT;
    int K = kt * 32 + 16 * (lane >> 4) + idx;
    int N = nt * 16 + (lane & 15);
    dst[e] = (_Float16)src[K * Ncols + N];
}

// 3) LN1 + remap NCHW -> token layout [bng][n][C] f16
__global__ void k_ln1(const float* __restrict__ xin, const float* __restrict__ gw,
                      const float* __restrict__ gb, _Float16* __restrict__ out) {
    int t = blockIdx.x * 256 + threadIdx.x;
    if (t >= TOKS) return;
    int b = t / HWp, rem = t % HWp;
    const float* p = xin + (size_t)b * Cd * HWp + rem;
    float s = 0.f, ss = 0.f;
    for (int c = 0; c < Cd; ++c) { float v = p[(size_t)c * HWp]; s += v; ss += v * v; }
    float mu = s * (1.f / 64.f);
    float inv = rsqrtf(ss * (1.f / 64.f) - mu * mu + 1e-6f);
    int y = rem / Wd, xx = rem % Wd;
    int hw = y >> 3, r = y & 7, wi = xx >> 3, c8 = xx & 7;
    int ghi = hw / 9, wgh = hw % 9, gwi = wi / 9, wgw = wi % 9;
    int bng = b * 16 + ghi * 4 + gwi;
    int n = (wgh * 9 + wgw) * 64 + r * 8 + c8;
    _Float16* o = out + ((size_t)bng * NPB + n) * Cd;
    for (int c = 0; c < Cd; ++c) {
        float v = p[(size_t)c * HWp];
        o[c] = (_Float16)((v - mu) * inv * gw[c] + gb[c]);
    }
}

// 4) window keys: mean over 64 tokens per window
__global__ void k_keys(const _Float16* __restrict__ xn, float* __restrict__ key) {
    int wi = blockIdx.x;           // bng*81 + wid
    int c  = threadIdx.x;          // 64 threads
    const _Float16* p = xn + (size_t)wi * 64 * Cd + c;
    float s = 0.f;
    for (int t = 0; t < 64; ++t) s += (float)p[t * Cd];
    key[(size_t)wi * Cd + c] = s * (1.f / 64.f);
}

// 5) assign = argmax_k <token, key_k>
__global__ void __launch_bounds__(288) k_assign(const _Float16* __restrict__ xn,
                                                const float* __restrict__ key,
                                                int* __restrict__ assign) {
    __shared__ float lk[NCH * Cd];
    int bng = blockIdx.x / 18;
    int blkin = blockIdx.x % 18;
    for (int i = threadIdx.x; i < NCH * Cd; i += 288)
        lk[i] = key[(size_t)bng * NCH * Cd + i];
    __syncthreads();
    int tl = blkin * 288 + threadIdx.x;
    const _Float16* row = xn + ((size_t)bng * NPB + tl) * Cd;
    float xr[Cd];
    for (int c = 0; c < Cd; ++c) xr[c] = (float)row[c];
    float best = -1e30f; int bi = 0;
    for (int k = 0; k < NCH; ++k) {
        float d = 0.f;
        for (int c = 0; c < Cd; ++c) d += xr[c] * lk[k * Cd + c];
        if (d > best) { best = d; bi = k; }
    }
    assign[(size_t)bng * NPB + tl] = bi;
}

// 6) stable counting sort per bng (== stable argsort of assign)
__global__ void __launch_bounds__(256) k_sort(const int* __restrict__ assign,
                                              int* __restrict__ sidx,
                                              int* __restrict__ ids) {
    __shared__ unsigned short ph[256 * NCH];
    __shared__ int binBase[NCH];
    __shared__ int binCnt[NCH];
    int bng = blockIdx.x, t = threadIdx.x;
    const int* a = assign + (size_t)bng * NPB;
    for (int b = 0; b < NCH; ++b) ph[t * NCH + b] = 0;
    int n0 = t * 21;
    for (int i = 0; i < 21; ++i) {
        int n = n0 + i;
        if (n < NPB) ph[t * NCH + a[n]]++;
    }
    __syncthreads();
    if (t < NCH) {
        int tot = 0;
        for (int tt = 0; tt < 256; ++tt) {
            unsigned short c = ph[tt * NCH + t];
            ph[tt * NCH + t] = (unsigned short)tot;
            tot += c;
        }
        binCnt[t] = tot;
    }
    __syncthreads();
    if (t == 0) {
        int base = 0;
        for (int b = 0; b < NCH; ++b) { binBase[b] = base; base += binCnt[b]; }
    }
    __syncthreads();
    for (int i = 0; i < 21; ++i) {
        int n = n0 + i;
        if (n >= NPB) break;
        int b = a[n];
        int pos = binBase[b] + ph[t * NCH + b];
        ph[t * NCH + b]++;
        sidx[(size_t)bng * NPB + pos] = n;
        ids [(size_t)bng * NPB + pos] = b;
    }
}

// 7) gather x_sorted (16B per thread)
__global__ void k_gather(const _Float16* __restrict__ xn, const int* __restrict__ sidx,
                         _Float16* __restrict__ xs) {
    int tid = blockIdx.x * 256 + threadIdx.x;
    if (tid >= TOKS * 8) return;
    int s = tid >> 3, part = tid & 7;
    int bng = s / NPB;
    int src = sidx[s];
    const v8h* in  = (const v8h*)(xn + ((size_t)bng * NPB + src) * Cd) + part;
    v8h*       out = (v8h*)(xs + (size_t)s * Cd) + part;
    *out = *in;
}

// 8) fused q/k/v/gate GEMMs (M=128/block, 8 waves x 16 rows)
__global__ void __launch_bounds__(256) k_qkvg(
    const _Float16* __restrict__ xs,
    const _Float16* __restrict__ wq, const _Float16* __restrict__ wk,
    const _Float16* __restrict__ wv, const _Float16* __restrict__ wg,
    const float* __restrict__ bq, const float* __restrict__ bk,
    const float* __restrict__ bv, const float* __restrict__ bg,
    _Float16* __restrict__ q, _Float16* __restrict__ k,
    _Float16* __restrict__ v, _Float16* __restrict__ g) {
    int lane = threadIdx.x & 31, wave = threadIdx.x >> 5;
    int row0 = blockIdx.x * 128 + wave * 16;
    const _Float16* arow = xs + (size_t)row0 * Cd;
    v16h a0 = frag_a_rm(arow, Cd, 0, lane);
    v16h a1 = frag_a_rm(arow, Cd, 32, lane);
    int col16 = lane & 15, hf = lane >> 4;
    const _Float16* wmats[4] = { wq, wk, wv, wg };
    const float*    bias[4]  = { bq, bk, bv, bg };
    _Float16*       outs[4]  = { q, k, v, g };
#pragma unroll
    for (int m = 0; m < 4; ++m) {
#pragma unroll
        for (int nt = 0; nt < 4; ++nt) {
            v8f acc = {};
            acc = wmma_f16(a0, frag_b_swz(wmats[m], nt,     lane), acc);
            acc = wmma_f16(a1, frag_b_swz(wmats[m], 4 + nt, lane), acc);
            int col = nt * 16 + col16;
            float bb = bias[m][col];
            _Float16* o = outs[m] + (size_t)row0 * Cd + col;
#pragma unroll
            for (int r = 0; r < 8; ++r) {
                float val = acc[r] + bb;
                if (m == 3) val = gelu_f(val);
                o[(size_t)(8 * hf + r) * Cd] = (_Float16)val;
            }
        }
    }
}

// 9) attention: S=QK^T, mask, softmax, O=PV, gate, proj, scatter x2=x1+Y
__global__ void __launch_bounds__(128) k_attn(
    const _Float16* __restrict__ q, const _Float16* __restrict__ kk,
    const _Float16* __restrict__ vv, const _Float16* __restrict__ gg,
    const int* __restrict__ ids, const int* __restrict__ sidx,
    const _Float16* __restrict__ wp, const float* __restrict__ bp,
    const float* __restrict__ x1, float* __restrict__ x2) {
    __shared__ _Float16 slab[4][16][128];
    int lane = threadIdx.x & 31, wave = threadIdx.x >> 5;
    int bng = blockIdx.x / NCH, ch = blockIdx.x % NCH;
    size_t tokbase = (size_t)bng * NPB;
    int qrow0 = ch * 64 + wave * 16;
    int col16 = lane & 15, hf = lane >> 4;

    const _Float16* arow = q + (tokbase + qrow0) * Cd;
    v16h aq0 = frag_a_rm(arow, Cd, 0, lane);
    v16h aq1 = frag_a_rm(arow, Cd, 32, lane);

    int qid[8];
#pragma unroll
    for (int r = 0; r < 8; ++r) qid[r] = ids[tokbase + qrow0 + 8 * hf + r];

    float sarr[8][8];
    int kvid[8];
#pragma unroll
    for (int nt = 0; nt < 8; ++nt) {
        int slot = nt * 16 + col16;
        int pos = ch * 64 - 32 + slot;
        bool valid = (pos >= 0) && (pos < NPB);
        v16h b0 = {}, b1 = {};
        if (valid) {
            const _Float16* krow = kk + (tokbase + pos) * Cd;
            b0 = frag_bT(krow, 0, lane);
            b1 = frag_bT(krow, 32, lane);
            kvid[nt] = ids[tokbase + pos];
        } else {
            kvid[nt] = -1;
        }
        v8f acc = {};
        acc = wmma_f16(aq0, b0, acc);
        acc = wmma_f16(aq1, b1, acc);
#pragma unroll
        for (int r = 0; r < 8; ++r) sarr[nt][r] = acc[r];
    }

    // masked softmax over 128 kv slots (8 reg tiles x 16 lanes in half-wave)
#pragma unroll
    for (int r = 0; r < 8; ++r) {
        float mx = -1e30f;
#pragma unroll
        for (int nt = 0; nt < 8; ++nt) {
            float l = (qid[r] == kvid[nt]) ? sarr[nt][r] * 0.125f : -10000.f;
            sarr[nt][r] = l;
            mx = fmaxf(mx, l);
        }
#pragma unroll
        for (int m = 1; m < 16; m <<= 1) mx = fmaxf(mx, __shfl_xor(mx, m, 32));
        float sm = 0.f;
#pragma unroll
        for (int nt = 0; nt < 8; ++nt) {
            float e = expf(sarr[nt][r] - mx);
            sarr[nt][r] = e;
            sm += e;
        }
#pragma unroll
        for (int m = 1; m < 16; m <<= 1) sm += __shfl_xor(sm, m, 32);
        float rinv = 1.f / sm;
#pragma unroll
        for (int nt = 0; nt < 8; ++nt) sarr[nt][r] *= rinv;
    }

    // P -> LDS (f16), each wave owns its own 16x128 slab
#pragma unroll
    for (int nt = 0; nt < 8; ++nt)
#pragma unroll
        for (int r = 0; r < 8; ++r)
            slab[wave][8 * hf + r][nt * 16 + col16] = (_Float16)sarr[nt][r];
    __syncthreads();

    // O = P @ V  (K=128, N=64)
    v16h ap[4];
#pragma unroll
    for (int ks = 0; ks < 4; ++ks)
        ap[ks] = frag_a_rm(&slab[wave][0][0], 128, ks * 32, lane);
    v8f oacc[4];
#pragma unroll
    for (int nt = 0; nt < 4; ++nt) oacc[nt] = (v8f){};
#pragma unroll
    for (int ks = 0; ks < 4; ++ks) {
#pragma unroll
        for (int nt = 0; nt < 4; ++nt) {
            v16h b;
#pragma unroll
            for (int i = 0; i < 16; ++i) {
                int slot = ks * 32 + 16 * hf + i;
                int pos = ch * 64 - 32 + slot;
                _Float16 val = (_Float16)0.f;
                if (pos >= 0 && pos < NPB)
                    val = vv[(tokbase + pos) * Cd + nt * 16 + col16];
                b[i] = val;
            }
            oacc[nt] = wmma_f16(ap[ks], b, oacc[nt]);
        }
    }

    // gate and stash gated context back into slab (first 64 cols)
    __syncthreads();
#pragma unroll
    for (int nt = 0; nt < 4; ++nt) {
        int col = nt * 16 + col16;
#pragma unroll
        for (int r = 0; r < 8; ++r) {
            int m = 8 * hf + r;
            float ga = (float)gg[(tokbase + qrow0 + m) * Cd + col];
            slab[wave][m][col] = (_Float16)(oacc[nt][r] * ga);
        }
    }
    __syncthreads();

    // proj GEMM (K=64) + scatter-add with shortcut into x2 (NCHW)
    v16h ag0 = frag_a_rm(&slab[wave][0][0], 128, 0, lane);
    v16h ag1 = frag_a_rm(&slab[wave][0][0], 128, 32, lane);
    int b_   = bng >> 4, gidx = bng & 15;
    int ghi  = gidx >> 2, gwi = gidx & 3;
#pragma unroll
    for (int nt = 0; nt < 4; ++nt) {
        v8f y = {};
        y = wmma_f16(ag0, frag_b_swz(wp, nt,     lane), y);
        y = wmma_f16(ag1, frag_b_swz(wp, 4 + nt, lane), y);
        int col = nt * 16 + col16;
        float bb = bp[col];
#pragma unroll
        for (int r = 0; r < 8; ++r) {
            int m = 8 * hf + r;
            int nloc = sidx[tokbase + qrow0 + m];
            int wid = nloc >> 6, p = nloc & 63;
            int wgh = wid / 9, wgw = wid % 9;
            int yy = (ghi * 9 + wgh) * 8 + (p >> 3);
            int xx = (gwi * 9 + wgw) * 8 + (p & 7);
            size_t oidx = (size_t)(b_ * Cd + col) * HWp + yy * Wd + xx;
            x2[oidx] = x1[oidx] + (y[r] + bb);
        }
    }
}

// 10) LN2 -> xn2 (f16, NHWC token order)
__global__ void k_ln2(const float* __restrict__ xin, const float* __restrict__ gw,
                      const float* __restrict__ gb, _Float16* __restrict__ out) {
    int t = blockIdx.x * 256 + threadIdx.x;
    if (t >= TOKS) return;
    int b = t / HWp, rem = t % HWp;
    const float* p = xin + (size_t)b * Cd * HWp + rem;
    float s = 0.f, ss = 0.f;
    for (int c = 0; c < Cd; ++c) { float v = p[(size_t)c * HWp]; s += v; ss += v * v; }
    float mu = s * (1.f / 64.f);
    float inv = rsqrtf(ss * (1.f / 64.f) - mu * mu + 1e-6f);
    _Float16* o = out + (size_t)t * Cd;
    for (int c = 0; c < Cd; ++c) {
        float v = p[(size_t)c * HWp];
        o[c] = (_Float16)((v - mu) * inv * gw[c] + gb[c]);
    }
}

// 11) fc1 + GELU -> h (f16, [token][128])
__global__ void __launch_bounds__(256) k_fc1(const _Float16* __restrict__ xn2,
                                             const _Float16* __restrict__ w1,
                                             const float* __restrict__ b1,
                                             _Float16* __restrict__ h) {
    int lane = threadIdx.x & 31, wave = threadIdx.x >> 5;
    int row0 = blockIdx.x * 128 + wave * 16;
    const _Float16* arow = xn2 + (size_t)row0 * Cd;
    v16h a0 = frag_a_rm(arow, Cd, 0, lane);
    v16h a1 = frag_a_rm(arow, Cd, 32, lane);
    int col16 = lane & 15, hf = lane >> 4;
#pragma unroll
    for (int nt = 0; nt < 8; ++nt) {
        v8f acc = {};
        acc = wmma_f16(a0, frag_b_swz(w1, nt,     lane), acc);
        acc = wmma_f16(a1, frag_b_swz(w1, 8 + nt, lane), acc);
        int col = nt * 16 + col16;
        float bb = b1[col];
#pragma unroll
        for (int r = 0; r < 8; ++r)
            h[((size_t)row0 + 8 * hf + r) * hCd + col] = (_Float16)gelu_f(acc[r] + bb);
    }
}

// 12) depthwise conv 5x5 on h (NHWC), hs = h + gelu(conv)
__global__ void k_dw5(const _Float16* __restrict__ h, const float* __restrict__ w,
                      const float* __restrict__ bias, _Float16* __restrict__ hs) {
    int gid = blockIdx.x * 256 + threadIdx.x;
    if (gid >= TOKS * hCd) return;
    int c = gid & 127;
    int t = gid >> 7;
    int b = t / HWp, rem = t % HWp;
    int y = rem / Wd, xx = rem % Wd;
    float acc = bias[c];
    const _Float16* base = h + (size_t)b * HWp * hCd + c;
    for (int dy = 0; dy < 5; ++dy) {
        int yy = y + dy - 2;
        if ((unsigned)yy >= (unsigned)Hh) continue;
        for (int dx = 0; dx < 5; ++dx) {
            int xc = xx + dx - 2;
            if ((unsigned)xc >= (unsigned)Wd) continue;
            acc += (float)base[((size_t)yy * Wd + xc) * hCd] * w[c * 25 + dy * 5 + dx];
        }
    }
    hs[gid] = (_Float16)((float)h[gid] + gelu_f(acc));
}

// 13) fc2 + residual -> d_out (NCHW f32)
__global__ void __launch_bounds__(256) k_fc2(const _Float16* __restrict__ hs,
                                             const _Float16* __restrict__ w2,
                                             const float* __restrict__ b2,
                                             const float* __restrict__ x2,
                                             float* __restrict__ out) {
    int lane = threadIdx.x & 31, wave = threadIdx.x >> 5;
    int row0 = blockIdx.x * 128 + wave * 16;
    const _Float16* arow = hs + (size_t)row0 * hCd;
    v16h a[4];
#pragma unroll
    for (int ks = 0; ks < 4; ++ks) a[ks] = frag_a_rm(arow, hCd, ks * 32, lane);
    int col16 = lane & 15, hf = lane >> 4;
#pragma unroll
    for (int nt = 0; nt < 4; ++nt) {
        v8f acc = {};
#pragma unroll
        for (int ks = 0; ks < 4; ++ks)
            acc = wmma_f16(a[ks], frag_b_swz(w2, ks * 4 + nt, lane), acc);
        int col = nt * 16 + col16;
        float bb = b2[col];
#pragma unroll
        for (int r = 0; r < 8; ++r) {
            int t = row0 + 8 * hf + r;
            int b = t / HWp, rem = t % HWp;
            size_t oi = (size_t)(b * Cd + col) * HWp + rem;
            out[oi] = x2[oi] + acc[r] + bb;
        }
    }
}

// ---------------------------------------------------------------- launch

extern "C" void kernel_launch(void* const* d_in, const int* in_sizes, int n_in,
                              void* d_out, int out_size, void* d_ws, size_t ws_size,
                              hipStream_t stream) {
    const float* x     = (const float*)d_in[0];
    const float* pe_w  = (const float*)d_in[1];
    const float* pe_b  = (const float*)d_in[2];
    const float* n1_w  = (const float*)d_in[3];
    const float* n1_b  = (const float*)d_in[4];
    const float* q_w   = (const float*)d_in[5];
    const float* q_b   = (const float*)d_in[6];
    const float* k_w   = (const float*)d_in[7];
    const float* k_b   = (const float*)d_in[8];
    const float* v_w   = (const float*)d_in[9];
    const float* v_b   = (const float*)d_in[10];
    const float* g_w   = (const float*)d_in[11];
    const float* g_b   = (const float*)d_in[12];
    const float* p_w   = (const float*)d_in[13];
    const float* p_b   = (const float*)d_in[14];
    const float* n2_w  = (const float*)d_in[15];
    const float* n2_b  = (const float*)d_in[16];
    const float* fc1_w = (const float*)d_in[17];
    const float* fc1_b = (const float*)d_in[18];
    const float* dw_w  = (const float*)d_in[19];
    const float* dw_b  = (const float*)d_in[20];
    const float* fc2_w = (const float*)d_in[21];
    const float* fc2_b = (const float*)d_in[22];

    char* ws = (char*)d_ws;
    const size_t SZ_F32 = (size_t)TOKS * Cd * 4;   // 84,934,656
    const size_t SZ_F16 = (size_t)TOKS * Cd * 2;   // 42,467,328
    size_t o = 0;
    float*    x1  = (float*)   (ws + o);               o += SZ_F32;
    size_t    oXN = o;
    _Float16* xn  = (_Float16*)(ws + o);               o += SZ_F16;
    _Float16* xs  = (_Float16*)(ws + o);               o += SZ_F16;
    size_t    oQ  = o;
    _Float16* qb  = (_Float16*)(ws + o);               o += SZ_F16;
    _Float16* kb  = (_Float16*)(ws + o);               o += SZ_F16;
    size_t    oV  = o;
    _Float16* vb  = (_Float16*)(ws + o);               o += SZ_F16;
    _Float16* gb  = (_Float16*)(ws + o);               o += SZ_F16;
    float*    key = (float*)   (ws + o);               o += (size_t)64 * NCH * Cd * 4;
    int*      asn = (int*)     (ws + o);               o += (size_t)TOKS * 4;
    int*      srt = (int*)     (ws + o);               o += (size_t)TOKS * 4;
    int*      idb = (int*)     (ws + o);               o += (size_t)TOKS * 4;
    _Float16* wqs = (_Float16*)(ws + o);               o += 8192;
    _Float16* wks = (_Float16*)(ws + o);               o += 8192;
    _Float16* wvs = (_Float16*)(ws + o);               o += 8192;
    _Float16* wgs = (_Float16*)(ws + o);               o += 8192;
    _Float16* wps = (_Float16*)(ws + o);               o += 8192;
    _Float16* w1s = (_Float16*)(ws + o);               o += 16384;
    _Float16* w2s = (_Float16*)(ws + o);               o += 16384;
    // aliases (buffers dead by the time these are written):
    float*    x2  = (float*)   (ws + oXN);   // overlays xn+xs
    _Float16* xn2 = (_Float16*)(ws);         // overlays x1
    _Float16* hb  = (_Float16*)(ws + oQ);    // overlays q+k
    _Float16* hsb = (_Float16*)(ws + oV);    // overlays v+g
    (void)ws_size; (void)n_in; (void)in_sizes; (void)out_size;

    // 1) pos-embed conv + residual
    k_pe<<<(Bn * Cd * HWp + 255) / 256, 256, 0, stream>>>(x, pe_w, pe_b, x1);
    // 2) weight swizzles
    k_wprep<<<16, 256, 0, stream>>>(q_w,   wqs, 64,  4, 4096);
    k_wprep<<<16, 256, 0, stream>>>(k_w,   wks, 64,  4, 4096);
    k_wprep<<<16, 256, 0, stream>>>(v_w,   wvs, 64,  4, 4096);
    k_wprep<<<16, 256, 0, stream>>>(g_w,   wgs, 64,  4, 4096);
    k_wprep<<<16, 256, 0, stream>>>(p_w,   wps, 64,  4, 4096);
    k_wprep<<<32, 256, 0, stream>>>(fc1_w, w1s, 128, 8, 8192);
    k_wprep<<<32, 256, 0, stream>>>(fc2_w, w2s, 64,  4, 8192);
    // 3) LN1 into token layout
    k_ln1<<<(TOKS + 255) / 256, 256, 0, stream>>>(x1, n1_w, n1_b, xn);
    // 4) window keys
    k_keys<<<64 * NCH, 64, 0, stream>>>(xn, key);
    // 5) cluster assignment
    k_assign<<<64 * 18, 288, 0, stream>>>(xn, key, asn);
    // 6) stable counting sort
    k_sort<<<64, 256, 0, stream>>>(asn, srt, idb);
    // 7) gather sorted tokens
    k_gather<<<(TOKS * 8 + 255) / 256, 256, 0, stream>>>(xn, srt, xs);
    // 8) q/k/v/gate projections (WMMA)
    k_qkvg<<<TOKS / 128, 256, 0, stream>>>(xs, wqs, wks, wvs, wgs,
                                           q_b, k_b, v_b, g_b, qb, kb, vb, gb);
    // 9) attention + gating + proj + scatter residual (WMMA)
    k_attn<<<64 * NCH, 128, 0, stream>>>(qb, kb, vb, gb, idb, srt, wps, p_b, x1, x2);
    // 10) LN2
    k_ln2<<<(TOKS + 255) / 256, 256, 0, stream>>>(x2, n2_w, n2_b, xn2);
    // 11) fc1 + GELU (WMMA)
    k_fc1<<<TOKS / 128, 256, 0, stream>>>(xn2, w1s, fc1_b, hb);
    // 12) depthwise 5x5 + GELU + add
    k_dw5<<<(TOKS * hCd + 255) / 256, 256, 0, stream>>>(hb, dw_w, dw_b, hsb);
    // 13) fc2 + residual -> out
    k_fc2<<<TOKS / 128, 256, 0, stream>>>(hsb, w2s, fc2_b, x2, (float*)d_out);
}
